// ScalePrediction_12532714569768
// MI455X (gfx1250) — matrix operand order repaired and checked by reading, not compiled, not run on hardware
//
#include <hip/hip_runtime.h>
#include <hip/hip_bf16.h>
#include <math.h>

typedef _Float16 v16h __attribute__((ext_vector_type(16)));
typedef _Float16 v8h  __attribute__((ext_vector_type(8)));
typedef float    v8f  __attribute__((ext_vector_type(8)));

namespace {
constexpr int Vv  = 3;    // views
constexpr int Bn  = 2;    // batch
constexpr int Cn  = 64;   // channels
constexpr int NSn = 32;   // scale hypotheses
constexpr int Hn  = 128;
constexpr int Wn  = 128;
constexpr int HW  = Hn * Wn;
constexpr int TILE  = 16;          // pixels per wave (one WMMA M dim)
constexpr int TILES = HW / TILE;   // 1024 tiles per (b,s)
constexpr int WPB   = 4;           // waves per block
}

// ---------- small affine (3x4, implicit bottom row 0 0 0 1) helpers ----------
__device__ inline void affine_inv(const float* M, float* O) {
  float a00=M[0],a01=M[1],a02=M[2],  t0=M[3];
  float a10=M[4],a11=M[5],a12=M[6],  t1=M[7];
  float a20=M[8],a21=M[9],a22=M[10], t2=M[11];
  float det = a00*(a11*a22-a12*a21) - a01*(a10*a22-a12*a20) + a02*(a10*a21-a11*a20);
  float id = 1.0f/det;
  float i00 =  (a11*a22-a12*a21)*id, i01 = -(a01*a22-a02*a21)*id, i02 =  (a01*a12-a02*a11)*id;
  float i10 = -(a10*a22-a12*a20)*id, i11 =  (a00*a22-a02*a20)*id, i12 = -(a00*a12-a02*a10)*id;
  float i20 =  (a10*a21-a11*a20)*id, i21 = -(a00*a21-a01*a20)*id, i22 =  (a00*a11-a01*a10)*id;
  O[0]=i00; O[1]=i01; O[2]=i02;  O[3]  = -(i00*t0 + i01*t1 + i02*t2);
  O[4]=i10; O[5]=i11; O[6]=i12;  O[7]  = -(i10*t0 + i11*t1 + i12*t2);
  O[8]=i20; O[9]=i21; O[10]=i22; O[11] = -(i20*t0 + i21*t1 + i22*t2);
}

__device__ inline void affine_mul(const float* A, const float* B, float* C) {
  #pragma unroll
  for (int r = 0; r < 3; ++r) {
    #pragma unroll
    for (int c = 0; c < 4; ++c) {
      float s = (c == 3) ? A[r*4+3] : 0.0f;
      #pragma unroll
      for (int k = 0; k < 3; ++k) s += A[r*4+k] * B[k*4+c];
      C[r*4+c] = s;
    }
  }
}

// ---------- kernel 0: zero the cost accumulators ----------
__global__ void zero_kernel(float* p, int n) {
  int i = blockIdx.x * blockDim.x + threadIdx.x;
  if (i < n) p[i] = 0.0f;
}

// ---------- kernel 1: per (view,batch) projection rot/trans (12 floats) ----------
__global__ void xform_kernel(const float* __restrict__ intr,
                             const float* __restrict__ c2w,
                             float* __restrict__ xf) {
  int t = threadIdx.x;
  if (t >= (Vv - 1) * Bn) return;
  int v = t / Bn + 1;
  int b = t % Bn;

  float src_c2w[12], ref_c2w[12], src_w2c[12], ref_w2c[12];
  #pragma unroll
  for (int i = 0; i < 12; ++i) {
    src_c2w[i] = c2w[(v * Bn + b) * 16 + i];   // rows 0..2 of 4x4 row-major
    ref_c2w[i] = c2w[(0 * Bn + b) * 16 + i];
  }
  affine_inv(src_c2w, src_w2c);
  affine_inv(ref_c2w, ref_w2c);

  float srcP[12], refP[12];
  #pragma unroll
  for (int r = 0; r < 3; ++r) {
    #pragma unroll
    for (int c = 0; c < 3; ++c) {
      srcP[r*4+c] = intr[(v * Bn + b) * 9 + r*3 + c];
      refP[r*4+c] = intr[(0 * Bn + b) * 9 + r*3 + c];
    }
    srcP[r*4+3] = src_w2c[r*4+3];
    refP[r*4+3] = ref_w2c[r*4+3];
  }
  float Mref[12], Minv[12], tmp[12], proj[12];
  affine_mul(refP, ref_w2c, Mref);   // ref_P @ ref_w2c
  affine_inv(Mref, Minv);
  affine_mul(src_w2c, Minv, tmp);    // src_w2c @ inv(...)
  affine_mul(srcP, tmp, proj);       // src_P @ ...
  #pragma unroll
  for (int i = 0; i < 12; ++i) xf[t * 12 + i] = proj[i];
}

// ---------- helpers to build WMMA operands from LDS-staged diffs ----------
__device__ inline v16h load_A(const _Float16* buf, int row, int hi, int kb) {
  // A 16x32 f16: lane<16 -> K {kb*32+0..7, kb*32+16..23}; lane>=16 -> {+8..15, +24..31}
  const v8h* lo = (const v8h*)(buf + row * Cn + kb * 32 + hi * 8);
  const v8h* up = (const v8h*)(buf + row * Cn + kb * 32 + 16 + hi * 8);
  v8h a = *lo, c = *up;
  v16h r;
  #pragma unroll
  for (int i = 0; i < 8; ++i) { r[i] = a[i]; r[i + 8] = c[i]; }
  return r;
}

__device__ inline v16h load_B(const _Float16* buf, int col, int hi, int kb) {
  // B 32x16 f16: lane n<16 -> col n, K {kb*32+0..7, kb*32+16..23}; lane n+16 -> col n, K {+8..15, +24..31}
  const v8h* lo = (const v8h*)(buf + col * Cn + kb * 32 + hi * 8);
  const v8h* up = (const v8h*)(buf + col * Cn + kb * 32 + 16 + hi * 8);
  v8h a = *lo, c = *up;
  v16h r;
  #pragma unroll
  for (int i = 0; i < 8; ++i) { r[i] = a[i]; r[i + 8] = c[i]; }
  return r;
}

__device__ inline float vsel8(v8f v, int i) {
  float r = v[0];
  if (i == 1) r = v[1];
  if (i == 2) r = v[2];
  if (i == 3) r = v[3];
  if (i == 4) r = v[4];
  if (i == 5) r = v[5];
  if (i == 6) r = v[6];
  if (i == 7) r = v[7];
  return r;
}

// ---------- kernel 2: main cost kernel ----------
__global__ void __launch_bounds__(WPB * 32)
cost_kernel(const float* __restrict__ feat,
            const float* __restrict__ depth_init,
            const float* __restrict__ scale_hypo,
            const float* __restrict__ xf,
            float* __restrict__ cost_acc) {
  __shared__ __align__(16) _Float16 lds[WPB][2][TILE * Cn];

  const int wave = threadIdx.x >> 5;
  const int lane = threadIdx.x & 31;
  const int gwave = blockIdx.x * WPB + wave;   // 0 .. B*NS*TILES-1

  const int tile = gwave % TILES;
  const int bs   = gwave / TILES;
  const int s    = bs % NSn;
  const int b    = bs / NSn;

  const int p   = lane & 15;     // pixel within tile (row of A, col of B)
  const int hi  = lane >> 4;     // lane half
  const int pix = tile * TILE + p;
  const int y   = pix >> 7;      // W = 128
  const int x   = pix & 127;

  const float dep = depth_init[b * HW + pix] * scale_hypo[b * NSn + s];

  // per-view bilinear corner offsets + weights (zeros padding)
  int   coff[2][4];
  float cwt[2][4];
  #pragma unroll
  for (int vv = 0; vv < 2; ++vv) {
    const float* T = xf + (vv * Bn + b) * 12;
    const float fx = (float)x, fy = (float)y;
    const float rx = T[0]*fx + T[1]*fy + T[2];
    const float ry = T[4]*fx + T[5]*fy + T[6];
    const float rz = T[8]*fx + T[9]*fy + T[10];
    const float px = rx * dep + T[3];
    const float py = ry * dep + T[7];
    const float pz = rz * dep + T[11];
    const float u  = px / pz;
    const float w_ = py / pz;
    const float gx = u  / ((Wn - 1) * 0.5f) - 1.0f;
    const float gy = w_ / ((Hn - 1) * 0.5f) - 1.0f;
    const float ix = ((gx + 1.0f) * Wn - 1.0f) * 0.5f;
    const float iy = ((gy + 1.0f) * Hn - 1.0f) * 0.5f;
    const float x0f = floorf(ix), y0f = floorf(iy);
    const float x1f = x0f + 1.0f, y1f = y0f + 1.0f;
    const float wx1 = ix - x0f, wx0 = 1.0f - wx1;
    const float wy1 = iy - y0f, wy0 = 1.0f - wy1;
    const float cxs[4] = {x0f, x1f, x0f, x1f};
    const float cys[4] = {y0f, y0f, y1f, y1f};
    const float cws[4] = {wx0 * wy0, wx1 * wy0, wx0 * wy1, wx1 * wy1};
    #pragma unroll
    for (int k = 0; k < 4; ++k) {
      const bool valid = (cxs[k] >= 0.0f) && (cxs[k] <= (float)(Wn - 1)) &&
                         (cys[k] >= 0.0f) && (cys[k] <= (float)(Hn - 1));
      const int xi = (int)fminf(fmaxf(cxs[k], 0.0f), (float)(Wn - 1));
      const int yi = (int)fminf(fmaxf(cys[k], 0.0f), (float)(Hn - 1));
      coff[vv][k] = yi * Wn + xi;
      cwt[vv][k]  = valid ? cws[k] : 0.0f;
    }
  }

  const float* f0b = feat + (long)(0 * Bn + b) * Cn * HW;
  const float* f1b = feat + (long)(1 * Bn + b) * Cn * HW;
  const float* f2b = feat + (long)(2 * Bn + b) * Cn * HW;
  _Float16* buf1 = lds[wave][0];   // diff after view 1
  _Float16* buf2 = lds[wave][1];   // diff after view 2

  // 2 channels per iteration: lane (p, hi) handles channel ci+hi of pixel p
  for (int ci = 0; ci < Cn; ci += 2) {
    const int c = ci + hi;
    const long co = (long)c * HW;
    const float f0 = f0b[co + pix];
    const float w1 = cwt[0][0] * f1b[co + coff[0][0]] + cwt[0][1] * f1b[co + coff[0][1]]
                   + cwt[0][2] * f1b[co + coff[0][2]] + cwt[0][3] * f1b[co + coff[0][3]];
    const float w2 = cwt[1][0] * f2b[co + coff[1][0]] + cwt[1][1] * f2b[co + coff[1][1]]
                   + cwt[1][2] * f2b[co + coff[1][2]] + cwt[1][3] * f2b[co + coff[1][3]];
    const float d1 = f0 - w1;
    const float d2 = d1 - w2;
    buf1[p * Cn + c] = (_Float16)d1;
    buf2[p * Cn + c] = (_Float16)d2;
  }

  // wave-local LDS ordering: CDNA5 split counter wait (stores visible to loads)
  asm volatile("s_wait_dscnt 0" ::: "memory");

  // Gram diagonals via V_WMMA_F32_16X16X32_F16, two K=32 chunks for C=64
  v8f acc1 = {};
  v8f acc2 = {};
  #pragma unroll
  for (int kb = 0; kb < 2; ++kb) {
    acc1 = __builtin_amdgcn_wmma_f32_16x16x32_f16(
        false, load_A(buf1, p, hi, kb), false, load_B(buf1, p, hi, kb),
        (short)0, acc1, false, false);
    acc2 = __builtin_amdgcn_wmma_f32_16x16x32_f16(
        false, load_A(buf2, p, hi, kb), false, load_B(buf2, p, hi, kb),
        (short)0, acc2, false, false);
  }

  // diag(p,p): lanes 0..7 hold pixels 0..7 at acc[lane]; lanes 24..31 hold 8..15 at acc[lane-24]
  const bool act = (lane < 8) || (lane >= 24);
  const int ridx = (lane < 8) ? lane : (lane - 24);
  float cc = 0.0f;
  if (act) {
    const float n1 = vsel8(acc1, ridx & 7);
    const float n2 = vsel8(acc2, ridx & 7);
    cc = sqrtf(n1) + sqrtf(n2);
  }
  // wave reduction over the 16 diagonal carriers
  #pragma unroll
  for (int off = 16; off > 0; off >>= 1) cc += __shfl_xor(cc, off, 32);
  if (lane == 0) atomicAdd(&cost_acc[b * NSn + s], cc);
}

// ---------- kernel 3: normalize, softmax over NS, dot with scale_hypo ----------
__global__ void finalize_kernel(const float* __restrict__ cost_acc,
                                const float* __restrict__ scale_hypo,
                                float* __restrict__ out) {
  const int b = blockIdx.x;
  const int s = threadIdx.x;   // NS == 32 == wave32
  const float c = cost_acc[b * NSn + s] * (1.0f / ((Vv - 1) * (float)HW));
  float m = c;
  #pragma unroll
  for (int off = 16; off > 0; off >>= 1) m = fmaxf(m, __shfl_xor(m, off, 32));
  const float e = expf(c - m);
  float sum = e;
  #pragma unroll
  for (int off = 16; off > 0; off >>= 1) sum += __shfl_xor(sum, off, 32);
  float w = (e / sum) * scale_hypo[b * NSn + s];
  #pragma unroll
  for (int off = 16; off > 0; off >>= 1) w += __shfl_xor(w, off, 32);
  if (s == 0) out[b] = w;
}

extern "C" void kernel_launch(void* const* d_in, const int* in_sizes, int n_in,
                              void* d_out, int out_size, void* d_ws, size_t ws_size,
                              hipStream_t stream) {
  const float* feat = (const float*)d_in[0];   // (V,B,C,H,W)
  const float* intr = (const float*)d_in[1];   // (V,B,3,3)
  const float* c2w  = (const float*)d_in[2];   // (V,B,4,4)
  const float* sh   = (const float*)d_in[3];   // (B,NS)
  const float* dep  = (const float*)d_in[4];   // (B,H,W)
  float* out = (float*)d_out;                  // (B,)

  float* cost_acc = (float*)d_ws;              // B*NS floats
  float* xf = cost_acc + Bn * NSn;             // (V-1)*B*12 floats

  zero_kernel<<<1, 64, 0, stream>>>(cost_acc, Bn * NSn);
  xform_kernel<<<1, 32, 0, stream>>>(intr, c2w, xf);

  const int total_waves = Bn * NSn * TILES;    // 65536
  const int blocks = total_waves / WPB;        // 16384
  cost_kernel<<<blocks, WPB * 32, 0, stream>>>(feat, dep, sh, xf, cost_acc);

  finalize_kernel<<<Bn, 32, 0, stream>>>(cost_acc, sh, out);
}
